// SuperMacroGCN_20203526160737
// MI455X (gfx1250) — matrix operand order repaired and checked by reading, not compiled, not run on hardware
//
#include <hip/hip_runtime.h>

// ---------------------------------------------------------------------------
// GCN 3-layer pipeline for gfx1250 (MI455X), wave32.
//   h    = X @ W                     -> v_wmma_f32_16x16x32_bf16, fp32 accum
//   agg  = b + dinv^2*h[i] + sum_e dinv[r]*dinv[c]*h[c]   (f32 L2 atomics)
//   act  = relu(bn(agg))
// Weights are pre-packed once into WMMA B-fragment order (bf16) so the GEMM
// loads each fragment with two coalesced b128 loads instead of 16 b32 loads.
// ---------------------------------------------------------------------------

typedef __attribute__((ext_vector_type(16))) __bf16 v16bf;
typedef __attribute__((ext_vector_type(8)))  float  v8f;

// ---------------- degree / normalization ----------------

__global__ __launch_bounds__(256) void deg_init_kernel(float* __restrict__ deg, int N) {
  int i = blockIdx.x * 256 + threadIdx.x;
  if (i < N) deg[i] = 1.0f;  // self-loop contributes 1
}

__global__ __launch_bounds__(256) void deg_accum_kernel(const int* __restrict__ row,
                                                        float* __restrict__ deg, int E) {
  int e = blockIdx.x * 256 + threadIdx.x;
  if (e < E) unsafeAtomicAdd(&deg[row[e]], 1.0f);  // native global_atomic_add_f32
}

__global__ __launch_bounds__(256) void dinv_kernel(float* __restrict__ d, int N) {
  int i = blockIdx.x * 256 + threadIdx.x;
  if (i < N) d[i] = rsqrtf(d[i]);  // deg >= 1 always
}

// ---------------- weight pre-pack: fp32 [64,64] -> bf16 WMMA B fragments ----
// Fragment f = ks*4 + nt covers B tile K in [ks*32, ks*32+32), N in [nt*16, nt*16+16).
// ISA B layout (32x16 bf16): lane = N%16 (+16 for upper half), v16bf elem i -> K = khalf + (i&7) + 16*(i>>3), khalf = 0/8.

__global__ __launch_bounds__(256) void pack_w_kernel(const float* __restrict__ W,
                                                     __bf16* __restrict__ out) {
  int t = blockIdx.x * 256 + threadIdx.x;  // 8 frags * 32 lanes * 16 elems = 4096
  if (t >= 4096) return;
  int i    = t & 15;
  int lane = (t >> 4) & 31;
  int f    = t >> 9;
  int ks   = f >> 2;
  int nt   = f & 3;
  int khalf = (lane >= 16) ? 8 : 0;
  int ncol  = lane & 15;
  int k = ks * 32 + khalf + (i & 7) + ((i >> 3) << 4);
  out[t] = (__bf16)W[k * 64 + nt * 16 + ncol];
}

// ---------------- fused GEMM + self-loop/bias init --------------------------
// One wave = 16 rows x 64 cols: 8 v_wmma_f32_16x16x32_bf16.
// Writes H = X@W and G = bias + dinv^2 * H (accumulator init for the scatter).

__global__ __launch_bounds__(256) void gcn_gemm64_wmma(const float* __restrict__ X,
                                                       const __bf16* __restrict__ Wp,
                                                       const float* __restrict__ bias,
                                                       const float* __restrict__ dinv,
                                                       float* __restrict__ H,
                                                       float* __restrict__ G, int nrows) {
  const int wave = (int)((blockIdx.x * (unsigned)blockDim.x + threadIdx.x) >> 5);
  const int lane = (int)(threadIdx.x & 31u);
  const int row0 = wave << 4;
  if (row0 >= nrows) return;

  const int mloc   = lane & 15;
  const int hiHalf = lane >> 4;
  int m = row0 + mloc;
  if (m >= nrows) m = nrows - 1;  // only if nrows % 16 != 0

  // A layout (16x32 bf16): lanes 0-15 hold K{0..7,16..23}, lanes 16-31 K{8..15,24..31}.
  const int kbase = hiHalf ? 8 : 0;

  // All 8 pre-packed B fragments: 2 coalesced b128 loads each.
  const v16bf* Bp = (const v16bf*)Wp;
  v16bf bfrag[8];
#pragma unroll
  for (int f = 0; f < 8; ++f) bfrag[f] = Bp[f * 32 + lane];

  v8f acc[4] = {v8f{0.f}, v8f{0.f}, v8f{0.f}, v8f{0.f}};

#pragma unroll
  for (int ks = 0; ks < 2; ++ks) {
    const float* xr = X + (size_t)m * 64 + ks * 32 + kbase;
    const float4 p0 = *(const float4*)(xr + 0);
    const float4 p1 = *(const float4*)(xr + 4);
    const float4 q0 = *(const float4*)(xr + 16);
    const float4 q1 = *(const float4*)(xr + 20);
    v16bf a;
    a[0]  = (__bf16)p0.x;  a[1]  = (__bf16)p0.y;  a[2]  = (__bf16)p0.z;  a[3]  = (__bf16)p0.w;
    a[4]  = (__bf16)p1.x;  a[5]  = (__bf16)p1.y;  a[6]  = (__bf16)p1.z;  a[7]  = (__bf16)p1.w;
    a[8]  = (__bf16)q0.x;  a[9]  = (__bf16)q0.y;  a[10] = (__bf16)q0.z;  a[11] = (__bf16)q0.w;
    a[12] = (__bf16)q1.x;  a[13] = (__bf16)q1.y;  a[14] = (__bf16)q1.z;  a[15] = (__bf16)q1.w;
#pragma unroll
    for (int nt = 0; nt < 4; ++nt) {
      acc[nt] = __builtin_amdgcn_wmma_f32_16x16x32_bf16(
          false, a, false, bfrag[ks * 4 + nt], (short)0, acc[nt], false, false);
    }
  }

  // C/D layout: VGPR i -> M = i (lanes 0-15) / 8+i (lanes 16-31), N = lane&15.
  const int mOff = hiHalf ? 8 : 0;
  float dsq[8];
#pragma unroll
  for (int i = 0; i < 8; ++i) {
    int rr = row0 + mOff + i;
    float d = dinv[rr < nrows ? rr : (nrows - 1)];
    dsq[i] = d * d;
  }
#pragma unroll
  for (int nt = 0; nt < 4; ++nt) {
    const int  colc = nt * 16 + mloc;
    const float bc  = bias[colc];
#pragma unroll
    for (int i = 0; i < 8; ++i) {
      const int rr = row0 + mOff + i;
      if (rr < nrows) {
        const size_t idx = (size_t)rr * 64 + colc;
        const float h = acc[nt][i];
        H[idx] = h;
        G[idx] = bc + dsq[i] * h;  // self-loop + bias accumulator init
      }
    }
  }
}

// ---------------- edge scatter ----------------
// 16 lanes per edge, float4 each: 256B coalesced gather of h[col], scaled f32
// atomic scatter into out[row]. h + out both fit in the 192MB L2.

__global__ __launch_bounds__(256) void edge_scatter_kernel(const float* __restrict__ H,
                                                           const float* __restrict__ dinv,
                                                           const int* __restrict__ row,
                                                           const int* __restrict__ col,
                                                           float* __restrict__ out, int E) {
  long long t = (long long)blockIdx.x * 256 + threadIdx.x;
  int e = (int)(t >> 4);
  if (e >= E) return;
  int l4 = (int)(t & 15) * 4;
  int r = row[e];
  int c = col[e];
  float w = dinv[r] * dinv[c];
  const float4 v = *(const float4*)(H + (size_t)c * 64 + l4);
  float* o = out + (size_t)r * 64 + l4;
  unsafeAtomicAdd(o + 0, v.x * w);
  unsafeAtomicAdd(o + 1, v.y * w);
  unsafeAtomicAdd(o + 2, v.z * w);
  unsafeAtomicAdd(o + 3, v.w * w);
}

// ---------------- batchnorm(eval) + relu ----------------

__global__ __launch_bounds__(256) void bn_relu_kernel(const float* __restrict__ agg,
                                                      const float* __restrict__ g,
                                                      const float* __restrict__ be,
                                                      const float* __restrict__ m,
                                                      const float* __restrict__ v,
                                                      float* __restrict__ out, int N) {
  long long t = (long long)blockIdx.x * 256 + threadIdx.x;
  if (t >= (long long)N * 64) return;
  int ch = (int)(t & 63);
  float x = (agg[t] - m[ch]) * rsqrtf(v[ch] + 1e-5f) * g[ch] + be[ch];
  out[t] = fmaxf(x, 0.0f);
}

// ---------------- host-side launch ----------------

static inline int cdiv_ll(long long a, long long b) { return (int)((a + b - 1) / b); }

extern "C" void kernel_launch(void* const* d_in, const int* in_sizes, int n_in,
                              void* d_out, int out_size, void* d_ws, size_t ws_size,
                              hipStream_t stream) {
  const float* x   = (const float*)d_in[0];
  const int* eidx  = (const int*)d_in[1];
  const float* W1  = (const float*)d_in[2];
  const float* b1  = (const float*)d_in[3];
  const float* g1  = (const float*)d_in[4];
  const float* be1 = (const float*)d_in[5];
  const float* m1  = (const float*)d_in[6];
  const float* v1  = (const float*)d_in[7];
  const float* W2  = (const float*)d_in[8];
  const float* b2  = (const float*)d_in[9];
  const float* g2  = (const float*)d_in[10];
  const float* be2 = (const float*)d_in[11];
  const float* m2  = (const float*)d_in[12];
  const float* v2  = (const float*)d_in[13];
  const float* W3  = (const float*)d_in[14];
  const float* b3  = (const float*)d_in[15];

  const int N = in_sizes[0] / 64;
  const int E = in_sizes[1] / 2;
  const int* row = eidx;      // segment ids (destination)
  const int* col = eidx + E;  // gather ids  (source)

  // workspace: D[N] | H[N*64] | G[N*64] | Wp1|Wp2|Wp3 (bf16, 4096 each)
  float* D = (float*)d_ws;
  size_t offH = ((size_t)N + 63) & ~(size_t)63;
  float* H = D + offH;
  float* G = H + (size_t)N * 64;
  __bf16* Wp1 = (__bf16*)(G + (size_t)N * 64);
  __bf16* Wp2 = Wp1 + 4096;
  __bf16* Wp3 = Wp2 + 4096;

  float* emb  = (float*)d_out;                  // [N,64] global_embeddings
  float* pred = (float*)d_out + (size_t)N * 64; // [N,64] hc_predictions

  const int gN   = cdiv_ll(N, 256);
  const int gE   = cdiv_ll(E, 256);
  const int gNC  = cdiv_ll((long long)N * 64, 256);
  const int gE16 = cdiv_ll((long long)E * 16, 256);
  const int gGemm = cdiv_ll(cdiv_ll(N, 16), 8);  // 8 waves/block, 16 rows/wave
  const int gPack = cdiv_ll(4096, 256);

  // ---- normalization: dinv = rsqrt(deg + selfloop) ----
  deg_init_kernel<<<gN, 256, 0, stream>>>(D, N);
  deg_accum_kernel<<<gE, 256, 0, stream>>>(row, D, E);
  dinv_kernel<<<gN, 256, 0, stream>>>(D, N);

  // ---- pre-pack weights into WMMA B-fragment order (bf16) ----
  pack_w_kernel<<<gPack, 256, 0, stream>>>(W1, Wp1);
  pack_w_kernel<<<gPack, 256, 0, stream>>>(W2, Wp2);
  pack_w_kernel<<<gPack, 256, 0, stream>>>(W3, Wp3);

  // ---- layer 1 ----
  gcn_gemm64_wmma<<<gGemm, 256, 0, stream>>>(x, Wp1, b1, D, H, G, N);   // h1->H, init->G
  edge_scatter_kernel<<<gE16, 256, 0, stream>>>(H, D, row, col, G, E);  // agg1 in G
  bn_relu_kernel<<<gNC, 256, 0, stream>>>(G, g1, be1, m1, v1, H, N);    // act1 -> H

  // ---- layer 2 ----
  gcn_gemm64_wmma<<<gGemm, 256, 0, stream>>>(H, Wp2, b2, D, G, pred, N); // h2->G, init->pred
  edge_scatter_kernel<<<gE16, 256, 0, stream>>>(G, D, row, col, pred, E); // agg2 in pred
  bn_relu_kernel<<<gNC, 256, 0, stream>>>(pred, g2, be2, m2, v2, emb, N); // embeddings

  // ---- layer 3 ----
  gcn_gemm64_wmma<<<gGemm, 256, 0, stream>>>(emb, Wp3, b3, D, H, pred, N); // h3->H, init->pred
  edge_scatter_kernel<<<gE16, 256, 0, stream>>>(H, D, row, col, pred, E);  // predictions
}
